// SmearAssociativeLM_34368328303131
// MI455X (gfx1250) — matrix operand order repaired and checked by reading, not compiled, not run on hardware
//
#include <hip/hip_runtime.h>
#include <math.h>

// ---------------------------------------------------------------------------
// Problem constants (match reference)
// ---------------------------------------------------------------------------
static constexpr int Vv = 32000;
static constexpr int Ee = 512;
static constexpr int Hh = 1024;
static constexpr int Mm = 256;
static constexpr int Bb = 2;
static constexpr int Ss = 1024;
static constexpr int GCc = 24;

// ---------------------------------------------------------------------------
// types / bf16 helpers
// ---------------------------------------------------------------------------
typedef __bf16 bf16_t;
typedef __attribute__((ext_vector_type(16))) bf16_t   v16bf;
typedef __attribute__((ext_vector_type(8)))  float    v8f;
typedef __attribute__((ext_vector_type(8)))  unsigned v8u;
typedef __attribute__((ext_vector_type(4)))  unsigned v4u;

__device__ __forceinline__ unsigned short bfbits(float f) {
  unsigned u = __builtin_bit_cast(unsigned, f);
  unsigned r = u + 0x7FFFu + ((u >> 16) & 1u);
  return (unsigned short)(r >> 16);
}
__device__ __forceinline__ bf16_t f2bf(float f) {
  unsigned short s = bfbits(f);
  return __builtin_bit_cast(bf16_t, s);
}
__device__ __forceinline__ float bf2f(bf16_t b) {
  unsigned short s = __builtin_bit_cast(unsigned short, b);
  unsigned u = ((unsigned)s) << 16;
  return __builtin_bit_cast(float, u);
}
__device__ __forceinline__ unsigned pack2bf(float lo, float hi) {
  return (unsigned)bfbits(lo) | ((unsigned)bfbits(hi) << 16);
}

// Fragment loaders. rowbase points at element 0 of this lane's 32-wide K slab
// (16B aligned). A-frag per lane = K runs [8h..8h+7] and [16+8h..23+8h];
// B-frag per lane = same K runs [8h..8h+7] and [16+8h..23+8h]. Both = two 16-byte chunks.
__device__ __forceinline__ v16bf frag_a(const void* rowbase, int half) {
  const v4u* p = (const v4u*)((const char*)rowbase + half * 16);
  v4u lo = p[0], hi = p[2];   // +0 bytes and +32 bytes
  v8u u;
  u[0] = lo[0]; u[1] = lo[1]; u[2] = lo[2]; u[3] = lo[3];
  u[4] = hi[0]; u[5] = hi[1]; u[6] = hi[2]; u[7] = hi[3];
  return __builtin_bit_cast(v16bf, u);
}
__device__ __forceinline__ v16bf frag_b(const void* rowbase, int half) {
  const v4u* p = (const v4u*)((const char*)rowbase + half * 16);
  v4u lo = p[0], hi = p[2];
  v8u u;
  u[0] = lo[0]; u[1] = lo[1]; u[2] = lo[2]; u[3] = lo[3];
  u[4] = hi[0]; u[5] = hi[1]; u[6] = hi[2]; u[7] = hi[3];
  return __builtin_bit_cast(v16bf, u);
}

// ---------------------------------------------------------------------------
// Generic WMMA GEMM:  C[b] = alpha * (A[b] @ W[b]^T) + bias
//   A: (M,K) row-major fp32/bf16;  W: (N,K) row-major fp32/bf16
// Block = 256 threads = 8 waves -> 128x64 C tile; wave w owns rows w*16..+16
// and all 4 N-subtiles (acc[4]); A-frag reused across 4 WMMAs per k-step.
// ---------------------------------------------------------------------------
template <bool A_BF16, bool W_BF16>
__global__ void wmma_gemm_bias(const void* __restrict__ Aptr,
                               const void* __restrict__ Wptr,
                               const float* __restrict__ bias,
                               float* __restrict__ Cf,
                               bf16_t* __restrict__ Cb,
                               int Mdim, int N, int K, float alpha,
                               long long strideA, long long strideW,
                               long long strideC) {
  __shared__ __attribute__((aligned(16))) bf16_t As[128][32];  // 8 KB
  __shared__ __attribute__((aligned(16))) bf16_t Wsh[64][32];  // 4 KB

  const int bz   = blockIdx.z;
  const int m0   = blockIdx.y * 128;
  const int n0   = blockIdx.x * 64;
  const int tid  = threadIdx.x;
  const int wave = tid >> 5;
  const int lane = tid & 31;
  const int half = lane >> 4;
  const int l16  = lane & 15;

  const float*  Af = (const float*)Aptr + bz * strideA;
  const bf16_t* Ab = (const bf16_t*)Aptr + bz * strideA;
  const float*  Wf = (const float*)Wptr + bz * strideW;
  const bf16_t* Wb = (const bf16_t*)Wptr + bz * strideW;

  v8f acc0 = {}, acc1 = {}, acc2 = {}, acc3 = {};

  for (int k0 = 0; k0 < K; k0 += 32) {
    // ---- stage A tile 128x32 (16 elements / thread)
    {
      int r = tid >> 1;
      int c = (tid & 1) << 4;
      long long g = (long long)(m0 + r) * K + (k0 + c);
      unsigned* dst = (unsigned*)&As[r][c];
      if (A_BF16) {
        const unsigned* src = (const unsigned*)(Ab + g);
#pragma unroll
        for (int i = 0; i < 8; ++i) dst[i] = src[i];
        if (k0 + 32 < K) __builtin_prefetch(Ab + g + 32, 0, 1);
      } else {
#pragma unroll
        for (int i = 0; i < 8; ++i)
          dst[i] = pack2bf(Af[g + 2 * i], Af[g + 2 * i + 1]);
        if (k0 + 32 < K) __builtin_prefetch(Af + g + 32, 0, 1);
      }
    }
    // ---- stage W tile 64x32 (8 elements / thread)
    {
      int idx = tid << 3;
      int r = idx >> 5;            // tid / 4
      int c = idx & 31;            // (tid % 4) * 8
      long long g = (long long)(n0 + r) * K + (k0 + c);
      unsigned* dst = (unsigned*)&Wsh[r][c];
      if (W_BF16) {
        const unsigned* src = (const unsigned*)(Wb + g);
#pragma unroll
        for (int i = 0; i < 4; ++i) dst[i] = src[i];
        if (k0 + 32 < K) __builtin_prefetch(Wb + g + 32, 0, 1);
      } else {
#pragma unroll
        for (int i = 0; i < 4; ++i)
          dst[i] = pack2bf(Wf[g + 2 * i], Wf[g + 2 * i + 1]);
        if (k0 + 32 < K) __builtin_prefetch(Wf + g + 32, 0, 1);
      }
    }
    __syncthreads();

    v16bf af = frag_a(&As[(wave << 4) + l16][0], half);
    v16bf b0 = frag_b(&Wsh[l16][0], half);
    acc0 = __builtin_amdgcn_wmma_f32_16x16x32_bf16(false, af, false, b0,
                                                   (short)0, acc0, false, false);
    v16bf b1 = frag_b(&Wsh[16 + l16][0], half);
    acc1 = __builtin_amdgcn_wmma_f32_16x16x32_bf16(false, af, false, b1,
                                                   (short)0, acc1, false, false);
    v16bf b2 = frag_b(&Wsh[32 + l16][0], half);
    acc2 = __builtin_amdgcn_wmma_f32_16x16x32_bf16(false, af, false, b2,
                                                   (short)0, acc2, false, false);
    v16bf b3 = frag_b(&Wsh[48 + l16][0], half);
    acc3 = __builtin_amdgcn_wmma_f32_16x16x32_bf16(false, af, false, b3,
                                                   (short)0, acc3, false, false);
    __syncthreads();
  }

  // epilogue: D layout row = v + 8*half, col = 16*j + lane%16
  const v8f* accs[4] = {&acc0, &acc1, &acc2, &acc3};
#pragma unroll
  for (int j = 0; j < 4; ++j) {
    const int col = n0 + (j << 4) + l16;
    const float bv = bias ? bias[col] : 0.0f;
    const v8f a = *accs[j];
#pragma unroll
    for (int v = 0; v < 8; ++v) {
      int row = m0 + (wave << 4) + (half << 3) + v;
      float out = a[v] * alpha + bv;
      long long ci = bz * strideC + (long long)row * N + col;
      if (Cf) Cf[ci] = out;
      if (Cb) Cb[ci] = f2bf(out);
    }
  }
}

// ---------------------------------------------------------------------------
// fp32 -> bf16 bulk conversion
// ---------------------------------------------------------------------------
__global__ void cvt_f32_bf16(const float* __restrict__ in,
                             bf16_t* __restrict__ out, long long n) {
  long long i = (long long)blockIdx.x * blockDim.x + threadIdx.x;
  if (i < n) out[i] = f2bf(in[i]);
}

// ---------------------------------------------------------------------------
// Embedding gather + smear (one wave per (b,t) row)
// ---------------------------------------------------------------------------
__global__ void embed_smear_kernel(const int* __restrict__ ids,
                                   const float* __restrict__ emb,
                                   const float* __restrict__ smearW,
                                   const float* __restrict__ smearLam,
                                   float* __restrict__ x) {
  int row = blockIdx.x * 8 + (threadIdx.x >> 5);
  int lane = threadIdx.x & 31;
  if (row >= Bb * Ss) return;
  int s = row & (Ss - 1);
  int id = ids[row];
  const float* er = emb + (long long)id * Ee;
  float partial = (lane < GCc) ? er[lane] * smearW[lane] : 0.0f;
#pragma unroll
  for (int off = 16; off > 0; off >>= 1)
    partial += __shfl_xor(partial, off, 32);
  float sg = 0.0f;
  const float* ep = er;
  if (s > 0) {
    sg = smearLam[0] * (1.0f / (1.0f + expf(-partial)));
    ep = emb + (long long)ids[row - 1] * Ee;
  }
  float* xr = x + (long long)row * Ee;
  for (int e = lane; e < Ee; e += 32) xr[e] = er[e] + sg * ep[e];
}

// ---------------------------------------------------------------------------
// GRU scan (sequential over S). One persistent 512-thread workgroup (16
// waves). Per step hg = W_hh(bf16) @ h via WMMA. Each wave processes 3 groups
// of 4 row-tiles: one B-frag load feeds 4 WMMAs (manual reuse, bounded
// registers — no LICM spill). h lives in a 16-row padded LDS panel so all
// lanes load unconditionally; per-thread hidden state stays in registers.
// ---------------------------------------------------------------------------
__global__ void __launch_bounds__(512)
gru_scan_kernel(const float* __restrict__ xg,     // (B,S,3H), b_ih included
                const bf16_t* __restrict__ Whh,   // (3H,H) bf16
                const float* __restrict__ bhh,    // (3H)
                bf16_t* __restrict__ states) {    // (B,S,H) bf16
  __shared__ __attribute__((aligned(16))) bf16_t h_pad[16][Hh];  // 32 KB
  __shared__ float hg[3 * Hh][2];                                // 24 KB

  const int tid  = threadIdx.x;
  const int wave = tid >> 5;
  const int lane = tid & 31;
  const int half = lane >> 4;
  const int l16  = lane & 15;

  // zero the padded h panel once; rows >= Bb stay zero forever
  for (int i = tid; i < 16 * Hh / 2; i += blockDim.x)
    ((unsigned*)h_pad)[i] = 0u;

  // per-thread persistent hidden state: flat indices tid*4 .. tid*4+3
  const int flat0 = tid << 2;
  const int gb = flat0 >> 10;          // batch for this thread's 4 units
  const int gj = flat0 & (Hh - 1);     // first hidden index (multiple of 4)
  float hreg[4] = {0.f, 0.f, 0.f, 0.f};
  __syncthreads();

  for (int t = 0; t < Ss; ++t) {
    // ---- Phase A: hg = Whh @ h (+ bhh) via WMMA -------------------------
#pragma unroll 1
    for (int grp = 0; grp < 3; ++grp) {
      asm volatile("" ::: "memory");  // stop LICM from hoisting h-panel loads
      // row-tiles rt_j = (grp*4 + j)*16 + wave, j = 0..3
      const int rtb = (grp << 2) * 16 + wave;
      const bf16_t* w0 = Whh + (long long)((rtb +  0) * 16 + l16) * Hh;
      const bf16_t* w1 = Whh + (long long)((rtb + 16) * 16 + l16) * Hh;
      const bf16_t* w2 = Whh + (long long)((rtb + 32) * 16 + l16) * Hh;
      const bf16_t* w3 = Whh + (long long)((rtb + 48) * 16 + l16) * Hh;
      v8f a0 = {}, a1 = {}, a2 = {}, a3 = {};
#pragma unroll 2
      for (int k0 = 0; k0 < Hh; k0 += 32) {
        v16bf bfrag = frag_b(&h_pad[l16][k0], half);
        v16bf f0 = frag_a(w0 + k0, half);
        a0 = __builtin_amdgcn_wmma_f32_16x16x32_bf16(false, f0, false, bfrag,
                                                     (short)0, a0, false, false);
        v16bf f1 = frag_a(w1 + k0, half);
        a1 = __builtin_amdgcn_wmma_f32_16x16x32_bf16(false, f1, false, bfrag,
                                                     (short)0, a1, false, false);
        v16bf f2 = frag_a(w2 + k0, half);
        a2 = __builtin_amdgcn_wmma_f32_16x16x32_bf16(false, f2, false, bfrag,
                                                     (short)0, a2, false, false);
        v16bf f3 = frag_a(w3 + k0, half);
        a3 = __builtin_amdgcn_wmma_f32_16x16x32_bf16(false, f3, false, bfrag,
                                                     (short)0, a3, false, false);
      }
      if (l16 < Bb) {
        const v8f* as[4] = {&a0, &a1, &a2, &a3};
#pragma unroll
        for (int j = 0; j < 4; ++j) {
          const v8f a = *as[j];
#pragma unroll
          for (int v = 0; v < 8; ++v) {
            int row = (rtb + j * 16) * 16 + (half << 3) + v;
            hg[row][l16] = a[v] + bhh[row];
          }
        }
      }
    }
    __syncthreads();

    // ---- Phase B: gates (4 consecutive hidden units per thread) ---------
    {
      const float* xgrow = xg + ((long long)gb * Ss + t) * (3 * Hh);
      float4 xr4 = *(const float4*)(xgrow + gj);
      float4 xz4 = *(const float4*)(xgrow + Hh + gj);
      float4 xn4 = *(const float4*)(xgrow + 2 * Hh + gj);
      float hn4[4];
#pragma unroll
      for (int e = 0; e < 4; ++e) {
        int j = gj + e;
        float xr = (&xr4.x)[e], xz = (&xz4.x)[e], xn = (&xn4.x)[e];
        float hr = hg[j][gb], hz = hg[Hh + j][gb], hn = hg[2 * Hh + j][gb];
        float r = 1.0f / (1.0f + expf(-(xr + hr)));
        float z = 1.0f / (1.0f + expf(-(xz + hz)));
        float n = tanhf(xn + r * hn);
        float hnew = (1.0f - z) * n + z * hreg[e];
        hreg[e] = hnew;
        hn4[e] = hnew;
      }
      unsigned p0 = pack2bf(hn4[0], hn4[1]);
      unsigned p1 = pack2bf(hn4[2], hn4[3]);
      unsigned* hp = (unsigned*)&h_pad[gb][gj];
      hp[0] = p0;
      hp[1] = p1;
      unsigned* sp = (unsigned*)(states + ((long long)gb * Ss + t) * Hh + gj);
      sp[0] = p0;
      sp[1] = p1;
    }
    __syncthreads();
  }
}

// ---------------------------------------------------------------------------
// g = sigmoid(states @ Wg^T + bg)  — GEMV, one wave per row
// ---------------------------------------------------------------------------
__global__ void gate_g_kernel(const bf16_t* __restrict__ states,
                              const float* __restrict__ Wg,
                              const float* __restrict__ bg,
                              float* __restrict__ g) {
  int row = blockIdx.x * 8 + (threadIdx.x >> 5);
  int lane = threadIdx.x & 31;
  if (row >= Bb * Ss) return;
  const bf16_t* sr = states + (long long)row * Hh;
  float acc = 0.0f;
  for (int i = lane; i < Hh; i += 32) acc += bf2f(sr[i]) * Wg[i];
#pragma unroll
  for (int off = 16; off > 0; off >>= 1) acc += __shfl_xor(acc, off, 32);
  if (lane == 0) g[row] = 1.0f / (1.0f + expf(-(acc + bg[0])));
}

// ---------------------------------------------------------------------------
// Causal softmax + gate scale, in place on scores (B,S,S).
// exp(finfo.min - max) == 0 exactly in fp32, so masked softmax + renorm
// reduces to causal softmax over t < s; row 0 -> zeros.
// ---------------------------------------------------------------------------
__global__ void softmax_gate_kernel(float* __restrict__ scores,
                                    const float* __restrict__ g,
                                    const float* __restrict__ memscale) {
  const int s = blockIdx.x, b = blockIdx.y;
  float* row = scores + ((long long)b * Ss + s) * Ss;
  __shared__ float red[8];
  const int tid = threadIdx.x, lane = tid & 31, wave = tid >> 5;

  float mx = -3.4e38f;
  for (int t2 = tid; t2 < s; t2 += 256) mx = fmaxf(mx, row[t2]);
#pragma unroll
  for (int off = 16; off > 0; off >>= 1) mx = fmaxf(mx, __shfl_xor(mx, off, 32));
  if (lane == 0) red[wave] = mx;
  __syncthreads();
  if (tid == 0) {
    float m = red[0];
#pragma unroll
    for (int i = 1; i < 8; ++i) m = fmaxf(m, red[i]);
    red[0] = m;
  }
  __syncthreads();
  mx = red[0];
  __syncthreads();

  float sum = 0.0f;
  for (int t2 = tid; t2 < s; t2 += 256) sum += expf(row[t2] - mx);
#pragma unroll
  for (int off = 16; off > 0; off >>= 1) sum += __shfl_xor(sum, off, 32);
  if (lane == 0) red[wave] = sum;
  __syncthreads();
  if (tid == 0) {
    float m = 0.0f;
#pragma unroll
    for (int i = 0; i < 8; ++i) m += red[i];
    red[0] = m;
  }
  __syncthreads();
  const float denom = red[0];

  const float scale =
      (s > 0) ? (g[b * Ss + s] * memscale[0]) / fmaxf(denom, 1e-30f) : 0.0f;
  for (int t2 = tid; t2 < Ss; t2 += 256)
    row[t2] = (t2 < s) ? expf(row[t2] - mx) * scale : 0.0f;
}

// ---------------------------------------------------------------------------
// Scatter-add: logits[b,s,ids[b,t]] += gated[b,s,t]  (t < s only; rest zero)
// ---------------------------------------------------------------------------
__global__ void scatter_add_kernel(const float* __restrict__ gated,
                                   const int* __restrict__ ids,
                                   float* __restrict__ logits) {
  int t2 = blockIdx.x * 256 + threadIdx.x;
  int s = blockIdx.y, b = blockIdx.z;
  if (t2 >= s) return;
  float v = gated[((long long)b * Ss + s) * Ss + t2];
  if (v != 0.0f)
    atomicAdd(&logits[((long long)b * Ss + s) * Vv + ids[b * Ss + t2]], v);
}

// ---------------------------------------------------------------------------
// kernel_launch
// ---------------------------------------------------------------------------
extern "C" void kernel_launch(void* const* d_in, const int* in_sizes, int n_in,
                              void* d_out, int out_size, void* d_ws,
                              size_t ws_size, hipStream_t stream) {
  const int*   ids       = (const int*)d_in[0];
  const float* emb       = (const float*)d_in[1];
  const float* W_ih      = (const float*)d_in[2];
  const float* W_hh      = (const float*)d_in[3];
  const float* b_ih      = (const float*)d_in[4];
  const float* b_hh      = (const float*)d_in[5];
  const float* Wq        = (const float*)d_in[6];
  const float* bq        = (const float*)d_in[7];
  const float* Wk        = (const float*)d_in[8];
  const float* bk        = (const float*)d_in[9];
  const float* Wg        = (const float*)d_in[10];
  const float* bg        = (const float*)d_in[11];
  const float* Whe       = (const float*)d_in[12];
  const float* bhe       = (const float*)d_in[13];
  const float* out_bias  = (const float*)d_in[14];
  const float* memscale  = (const float*)d_in[15];
  const float* smear_W   = (const float*)d_in[16];
  const float* smear_lam = (const float*)d_in[17];
  float* logits = (float*)d_out;

  // workspace carve-out
  char* wsp = (char*)d_ws;
  auto alloc = [&](size_t bytes) -> char* {
    char* p = wsp;
    wsp += (bytes + 255) & ~(size_t)255;
    return p;
  };
  float*  x         = (float*)alloc(sizeof(float) * Bb * Ss * Ee);
  float*  xg        = (float*)alloc(sizeof(float) * Bb * Ss * 3 * Hh);
  bf16_t* whh_bf    = (bf16_t*)alloc(sizeof(bf16_t) * 3 * Hh * Hh);
  bf16_t* states_bf = (bf16_t*)alloc(sizeof(bf16_t) * Bb * Ss * Hh);
  bf16_t* he_bf     = (bf16_t*)alloc(sizeof(bf16_t) * Bb * Ss * Ee);
  bf16_t* q_bf      = (bf16_t*)alloc(sizeof(bf16_t) * Bb * Ss * Mm);
  bf16_t* k_bf      = (bf16_t*)alloc(sizeof(bf16_t) * Bb * Ss * Mm);
  float*  scores    = (float*)alloc(sizeof(float) * Bb * Ss * Ss);
  float*  gvec      = (float*)alloc(sizeof(float) * Bb * Ss);
  bf16_t* emb_bf    = (bf16_t*)alloc(sizeof(bf16_t) * (long long)Vv * Ee);

  const dim3 blk(256);
  const int MR = Bb * Ss;  // 2048 flattened rows

  // 0) conversions
  {
    long long n = (long long)Vv * Ee;
    cvt_f32_bf16<<<dim3((unsigned)((n + 255) / 256)), blk, 0, stream>>>(emb, emb_bf, n);
    long long n2 = (long long)3 * Hh * Hh;
    cvt_f32_bf16<<<dim3((unsigned)((n2 + 255) / 256)), blk, 0, stream>>>(W_hh, whh_bf, n2);
  }
  // 1) embedding + smear
  embed_smear_kernel<<<dim3(MR / 8), blk, 0, stream>>>(ids, emb, smear_W, smear_lam, x);
  // 2) xg = x @ W_ih^T + b_ih   (2048 x 3072, K=512)
  wmma_gemm_bias<false, false><<<dim3(3 * Hh / 64, MR / 128, 1), blk, 0, stream>>>(
      x, W_ih, b_ih, xg, nullptr, MR, 3 * Hh, Ee, 1.0f, 0, 0, 0);
  // 3) GRU scan -> states (bf16)
  gru_scan_kernel<<<dim3(1), dim3(512), 0, stream>>>(xg, whh_bf, b_hh, states_bf);
  // 4) he = states @ Whe^T + bhe  (2048 x 512, K=1024) -> bf16
  wmma_gemm_bias<true, false><<<dim3(Ee / 64, MR / 128, 1), blk, 0, stream>>>(
      states_bf, Whe, bhe, nullptr, he_bf, MR, Ee, Hh, 1.0f, 0, 0, 0);
  // 5) q, k projections (2048 x 256, K=1024) -> bf16
  wmma_gemm_bias<true, false><<<dim3(Mm / 64, MR / 128, 1), blk, 0, stream>>>(
      states_bf, Wq, bq, nullptr, q_bf, MR, Mm, Hh, 1.0f, 0, 0, 0);
  wmma_gemm_bias<true, false><<<dim3(Mm / 64, MR / 128, 1), blk, 0, stream>>>(
      states_bf, Wk, bk, nullptr, k_bf, MR, Mm, Hh, 1.0f, 0, 0, 0);
  // 6) scores = (q @ k^T) / 16, batched over B (1024 x 1024, K=256)
  wmma_gemm_bias<true, true><<<dim3(Ss / 64, Ss / 128, Bb), blk, 0, stream>>>(
      q_bf, k_bf, nullptr, scores, nullptr, Ss, Ss, Mm, 0.0625f,
      (long long)Ss * Mm, (long long)Ss * Mm, (long long)Ss * Ss);
  // 7) g gate
  gate_g_kernel<<<dim3(MR / 8), blk, 0, stream>>>(states_bf, Wg, bg, gvec);
  // 8) causal softmax * (g * memory_scale), in place -> "gated"
  softmax_gate_kernel<<<dim3(Ss, Bb), blk, 0, stream>>>(scores, gvec, memscale);
  // 9) logits = he @ emb^T + out_bias  (2048 x 32000, K=512) — the big one
  wmma_gemm_bias<true, true><<<dim3(Vv / 64, MR / 128, 1), blk, 0, stream>>>(
      he_bf, emb_bf, out_bias, logits, nullptr, MR, Vv, Ee, 1.0f, 0, 0, 0);
  // 10) scatter-add gated attention into logits at token positions
  scatter_add_kernel<<<dim3(Ss / 256, Ss, Bb), blk, 0, stream>>>(scores, ids, logits);
}
